// ResidualBlock_755914244226
// MI455X (gfx1250) — matrix operand order repaired and checked by reading, not compiled, not run on hardware
//
#include <hip/hip_runtime.h>

typedef __attribute__((ext_vector_type(16))) _Float16 v16h;
typedef __attribute__((ext_vector_type(8)))  float    v8f;

#define Bn 4
#define Cin 64
#define Hh 128
#define Ww 128
#define HWn (Hh*Ww)
#define KKn 9
#define Ocn 64
#define KDIM (Cin*KKn)          // 576
#define KSTEPS (KDIM/32)        // 18
#define TILE_M 16
#define APAD 8
#define ASTRIDE (KDIM+APAD)     // 584 halves, keeps 16B alignment, breaks bank stride
#define OHALF 32

// -------- Kernel 1: offset/mask 3x3 conv (18 offset ch + 9 mask ch w/ sigmoid) ------
__global__ void offmask_kernel(const float* __restrict__ x,
                               const float* __restrict__ w_off, const float* __restrict__ b_off,
                               const float* __restrict__ w_mask, const float* __restrict__ b_mask,
                               float* __restrict__ off_out, float* __restrict__ msk_out) {
    int idx = blockIdx.x * blockDim.x + threadIdx.x;
    const int total = Bn * 27 * HWn;
    if (idx >= total) return;
    int pix = idx % HWn;
    int t   = idx / HWn;
    int ch  = t % 27;
    int b   = t / 27;
    int h = pix / Ww, w = pix % Ww;

    const float* wsel; float bias; int cc;
    if (ch < 18) { cc = ch;      wsel = w_off  + cc * Cin * 9; bias = b_off[cc]; }
    else         { cc = ch - 18; wsel = w_mask + cc * Cin * 9; bias = b_mask[cc]; }

    float acc = bias;
    const float* xb = x + (size_t)b * Cin * HWn;
    for (int c = 0; c < Cin; ++c) {
        const float* xc = xb + c * HWn;
        const float* wc = wsel + c * 9;
        #pragma unroll
        for (int ki = 0; ki < 3; ++ki) {
            int hy = h + ki - 1;
            if (hy < 0 || hy >= Hh) continue;
            #pragma unroll
            for (int kj = 0; kj < 3; ++kj) {
                int wx2 = w + kj - 1;
                if (wx2 < 0 || wx2 >= Ww) continue;
                acc = fmaf(xc[hy * Ww + wx2], wc[ki * 3 + kj], acc);
            }
        }
    }
    if (ch < 18) off_out[((size_t)b * 18 + cc) * HWn + pix] = acc;
    else         msk_out[((size_t)b * 9  + cc) * HWn + pix] = 1.0f / (1.0f + __expf(-acc));
}

// -------- Kernel 2: deformable sampling + implicit GEMM via v_wmma_f32_16x16x32_f16 --
__global__ void __launch_bounds__(64)
deform_wmma_kernel(const float* __restrict__ x, const float* __restrict__ off,
                   const float* __restrict__ msk, const float* __restrict__ wgt,
                   const float* __restrict__ resid, float* __restrict__ out) {
    __shared__ _Float16 As[TILE_M * ASTRIDE];   // 16*584*2 = 18,688 B
    __shared__ _Float16 Wt[OHALF  * ASTRIDE];   // 32*584*2 = 37,376 B   (total 56 KB)

    const int tid   = threadIdx.x;
    const int tile  = blockIdx.x;               // B*HW/16 tiles
    const int ohalf = blockIdx.y;               // 0..1 over O
    const int b     = tile / (HWn / TILE_M);
    const int pix0  = (tile % (HWn / TILE_M)) * TILE_M;

    // Stage weight half into LDS: Wt[o_local][kglob], kglob = c*9 + k
    for (int i = tid; i < OHALF * KDIM; i += 64) {
        int ol = i / KDIM, kg = i % KDIM;
        int c = kg / KKn, k = kg % KKn;
        int o = ohalf * OHALF + ol;
        float v = wgt[(((size_t)o * Cin + c) * 3 + (k / 3)) * 3 + (k % 3)];
        Wt[ol * ASTRIDE + kg] = (_Float16)v;
    }

    // Bilinear sampling into As[p][c*9+k], masked
    for (int pk = tid; pk < TILE_M * KKn; pk += 64) {
        int p = pk / KKn, k = pk % KKn;
        int pix = pix0 + p;
        int h = pix / Ww, w = pix % Ww;
        float dy = off[((size_t)b * 2 * KKn + k * 2 + 0) * HWn + pix];
        float dx = off[((size_t)b * 2 * KKn + k * 2 + 1) * HWn + pix];
        float m  = msk[((size_t)b * KKn + k) * HWn + pix];
        float yy = (float)(h - 1 + (k / 3)) + dy;
        float xx = (float)(w - 1 + (k % 3)) + dx;
        float y0f = floorf(yy), x0f = floorf(xx);
        int   y0 = (int)y0f,   x0 = (int)x0f;
        float wy = yy - y0f,   wx = xx - x0f;
        int   y1 = y0 + 1,     x1 = x0 + 1;
        bool vy0 = (y0 >= 0) && (y0 < Hh), vy1 = (y1 >= 0) && (y1 < Hh);
        bool vx0 = (x0 >= 0) && (x0 < Ww), vx1 = (x1 >= 0) && (x1 < Ww);
        int cy0 = min(max(y0, 0), Hh - 1), cy1 = min(max(y1, 0), Hh - 1);
        int cx0 = min(max(x0, 0), Ww - 1), cx1 = min(max(x1, 0), Ww - 1);
        float w00 = (vy0 && vx0) ? (1.f - wy) * (1.f - wx) * m : 0.f;
        float w01 = (vy0 && vx1) ? (1.f - wy) * wx         * m : 0.f;
        float w10 = (vy1 && vx0) ? wy         * (1.f - wx) * m : 0.f;
        float w11 = (vy1 && vx1) ? wy         * wx         * m : 0.f;
        int i00 = cy0 * Ww + cx0, i01 = cy0 * Ww + cx1;
        int i10 = cy1 * Ww + cx0, i11 = cy1 * Ww + cx1;
        const float* xb = x + (size_t)b * Cin * HWn;
        for (int c = 0; c < Cin; ++c) {
            const float* xc = xb + c * HWn;
            float s = w00 * xc[i00] + w01 * xc[i01] + w10 * xc[i10] + w11 * xc[i11];
            As[p * ASTRIDE + c * KKn + k] = (_Float16)s;
        }
    }
    __syncthreads();

    // Per-wave 16x16 N tile; K loop of 18 WMMAs (contiguous k = 16h + s order in both A and B)
    const int wv   = tid >> 5;          // 0..1
    const int lane = tid & 31;
    const int arow = lane & 15;
    const int akb  = (lane >> 4) << 4;  // A: lanes0-15 K 0..15, lanes16-31 K 16..31
    const int ncol = lane & 15;
    const int bkb  = (lane >> 4) << 4;  // B: lanes0-15 K 0..15, lanes16-31 K 16..31

    v8f acc = {0.f, 0.f, 0.f, 0.f, 0.f, 0.f, 0.f, 0.f};
    for (int ks = 0; ks < KSTEPS; ++ks) {
        const int kbase = ks * 32;
        v16h a, bfrag;
        const _Float16* ap = &As[arow * ASTRIDE + kbase + akb];
        #pragma unroll
        for (int e = 0; e < 16; ++e) a[e] = ap[e];
        const _Float16* bp = &Wt[(wv * 16 + ncol) * ASTRIDE + kbase + bkb];
        #pragma unroll
        for (int e = 0; e < 16; ++e) bfrag[e] = bp[e];
        acc = __builtin_amdgcn_wmma_f32_16x16x32_f16(
                  false, a, false, bfrag, (short)0, acc, false, false);
    }

    // C/D layout: lane n = lane&15 (channel), rows m = r + (lane>=16 ? 8 : 0)
    const int o     = ohalf * OHALF + wv * 16 + (lane & 15);
    const int mbase = (lane >> 4) << 3;
    float* ob = out + ((size_t)b * Ocn + o) * HWn;
    const float* rb = resid ? resid + ((size_t)b * Ocn + o) * HWn : nullptr;
    #pragma unroll
    for (int r = 0; r < 8; ++r) {
        int pix = pix0 + mbase + r;
        float v = acc[r];
        v = v > 0.f ? v : 0.f;          // ReLU
        if (rb) v += rb[pix];           // residual (+x) fused in block 2
        ob[pix] = v;
    }
}

extern "C" void kernel_launch(void* const* d_in, const int* in_sizes, int n_in,
                              void* d_out, int out_size, void* d_ws, size_t ws_size,
                              hipStream_t stream) {
    const float* x       = (const float*)d_in[0];
    const float* w_off1  = (const float*)d_in[1];
    const float* b_off1  = (const float*)d_in[2];
    const float* w_mask1 = (const float*)d_in[3];
    const float* b_mask1 = (const float*)d_in[4];
    const float* w1      = (const float*)d_in[5];
    const float* w_off2  = (const float*)d_in[6];
    const float* b_off2  = (const float*)d_in[7];
    const float* w_mask2 = (const float*)d_in[8];
    const float* b_mask2 = (const float*)d_in[9];
    const float* w2      = (const float*)d_in[10];

    float* ws      = (float*)d_ws;                      // needs ~24 MB
    float* off_buf = ws;                                // B*18*HW
    float* msk_buf = off_buf + (size_t)Bn * 18 * HWn;   // B*9*HW
    float* c1      = msk_buf + (size_t)Bn * 9  * HWn;   // B*64*HW
    float* outp    = (float*)d_out;

    dim3 b1(256), g1((Bn * 27 * HWn + 255) / 256);
    dim3 b2(64),  g2(Bn * HWn / TILE_M, 2);

    // Block 1: x -> c1
    offmask_kernel<<<g1, b1, 0, stream>>>(x, w_off1, b_off1, w_mask1, b_mask1, off_buf, msk_buf);
    deform_wmma_kernel<<<g2, b2, 0, stream>>>(x, off_buf, msk_buf, w1, nullptr, c1);
    // Block 2: c1 -> out (+x residual fused)
    offmask_kernel<<<g1, b1, 0, stream>>>(c1, w_off2, b_off2, w_mask2, b_mask2, off_buf, msk_buf);
    deform_wmma_kernel<<<g2, b2, 0, stream>>>(c1, off_buf, msk_buf, w2, x, outp);
}
